// MemoryRetriever_49392123904344
// MI455X (gfx1250) — matrix operand order repaired and checked by reading, not compiled, not run on hardware
//
#include <hip/hip_runtime.h>

typedef __attribute__((ext_vector_type(16))) __bf16 v16bf;
typedef __attribute__((ext_vector_type(8)))  __bf16 v8bf;
typedef __attribute__((ext_vector_type(8)))  float  v8f;

#define WMMA_BF16(a, b, c) \
  __builtin_amdgcn_wmma_f32_16x16x32_bf16(false, (a), false, (b), (short)0, (c), false, false)

// ---- WMMA operand loaders (bf16, wave32; contiguous k = 16*half + s in both A and B) ----
// A 16x32: lane holds row (lane&15); elems 0..15 = K[kb+half*16 .. +16)
//          (half = lane>>4)
static __device__ __forceinline__ v16bf ld_a16(const __bf16* p, int half) {
  const v8bf lo = *(const v8bf*)(p + half * 16);
  const v8bf hi = *(const v8bf*)(p + half * 16 + 8);
  v16bf r;
#pragma unroll
  for (int i = 0; i < 8; ++i) { r[i] = lo[i]; r[i + 8] = hi[i]; }
  return r;
}
// B 32x16: lane holds column (lane&15); 16 contiguous K starting at kb+half*16
static __device__ __forceinline__ v16bf ld_b16(const __bf16* p) {
  const v8bf lo = *(const v8bf*)(p);
  const v8bf hi = *(const v8bf*)(p + 8);
  v16bf r;
#pragma unroll
  for (int i = 0; i < 8; ++i) { r[i] = lo[i]; r[i + 8] = hi[i]; }
  return r;
}

// ---- Generic WMMA GEMM: out = epilogue(A[M,K] @ B[N,K]^T + bias) ----
// block = 256 thr (8 waves); wave tile 16x64; block tile 64x128.
// EPI: 0=f32, 1=bf16, 2=bf16*0.125, 3=sigmoid f32, 4=bf16 transposed store
template <int EPI>
__global__ __launch_bounds__(256) void gemm_bf16_kernel(
    const __bf16* __restrict__ A, int lda, const __bf16* __restrict__ B, int ldb,
    const float* __restrict__ bias, void* __restrict__ out, int ldo, int K) {
  const int wave = threadIdx.x >> 5;
  const int lane = threadIdx.x & 31;
  const int l15 = lane & 15, half = lane >> 4;
  const int m0 = blockIdx.y * 64 + (wave & 3) * 16;
  const int n0 = blockIdx.x * 128 + (wave >> 2) * 64;
  const __bf16* arow = A + (size_t)(m0 + l15) * lda;
  const __bf16* bp0 = B + (size_t)(n0 + 0 + l15) * ldb + half * 16;
  const __bf16* bp1 = B + (size_t)(n0 + 16 + l15) * ldb + half * 16;
  const __bf16* bp2 = B + (size_t)(n0 + 32 + l15) * ldb + half * 16;
  const __bf16* bp3 = B + (size_t)(n0 + 48 + l15) * ldb + half * 16;
  v8f acc0 = {}, acc1 = {}, acc2 = {}, acc3 = {};
  for (int kb = 0; kb < K; kb += 32) {
    const v16bf a = ld_a16(arow + kb, half);
    acc0 = WMMA_BF16(a, ld_b16(bp0 + kb), acc0);
    acc1 = WMMA_BF16(a, ld_b16(bp1 + kb), acc1);
    acc2 = WMMA_BF16(a, ld_b16(bp2 + kb), acc2);
    acc3 = WMMA_BF16(a, ld_b16(bp3 + kb), acc3);
  }
  v8f accs[4] = {acc0, acc1, acc2, acc3};
#pragma unroll
  for (int nt = 0; nt < 4; ++nt) {
    const int n = n0 + nt * 16 + l15;
    const float bs = bias[n];
#pragma unroll
    for (int r = 0; r < 8; ++r) {
      const int m = m0 + r + half * 8;
      const float x = accs[nt][r] + bs;
      if (EPI == 0)       ((float*)out)[(size_t)m * ldo + n] = x;
      else if (EPI == 1)  ((__bf16*)out)[(size_t)m * ldo + n] = (__bf16)x;
      else if (EPI == 2)  ((__bf16*)out)[(size_t)m * ldo + n] = (__bf16)(x * 0.125f);
      else if (EPI == 3)  ((float*)out)[(size_t)m * ldo + n] = 1.0f / (1.0f + __expf(-x));
      else                ((__bf16*)out)[(size_t)n * ldo + m] = (__bf16)x;  // transposed
    }
  }
}

// ---- Fused attention: 1 wave = 16 tokens x 1 head; 512 keys, HD=64 ----
__global__ __launch_bounds__(32) void attn_kernel(
    const __bf16* __restrict__ Q, const __bf16* __restrict__ Km,
    const __bf16* __restrict__ Vt, __bf16* __restrict__ ctx) {
  constexpr int SCP = 528;  // padded row stride (floats) to dodge bank conflicts
  __shared__ float sc[16 * SCP];
  const int h = blockIdx.y;
  const int t0 = blockIdx.x * 16;
  const int lane = threadIdx.x;
  const int l15 = lane & 15, half = lane >> 4;

  // Q tile (already scaled by 1/sqrt(HD)): two 16x32 A operands cover HD=64
  const __bf16* qrow = Q + (size_t)(t0 + l15) * 256 + h * 64;
  const v16bf a0 = ld_a16(qrow, half);
  const v16bf a1 = ld_a16(qrow + 32, half);

  // scores = Q @ K^T -> LDS
  for (int kc = 0; kc < 32; ++kc) {
    const int key = kc * 16 + l15;
    const __bf16* kr = Km + (size_t)key * 256 + h * 64 + half * 16;
    v8f s = {};
    s = WMMA_BF16(a0, ld_b16(kr), s);
    s = WMMA_BF16(a1, ld_b16(kr + 32), s);
#pragma unroll
    for (int r = 0; r < 8; ++r) sc[(r + half * 8) * SCP + kc * 16 + l15] = s[r];
  }
  __syncthreads();

  // softmax over 512 keys: lane handles row (lane&15), key half [half*256, +256)
  float* row = sc + l15 * SCP;
  const int kb0 = half * 256;
  float mx = -3.4e38f;
  for (int i = 0; i < 256; ++i) mx = fmaxf(mx, row[kb0 + i]);
  mx = fmaxf(mx, __shfl_xor(mx, 16, 32));
  float sum = 0.f;
  for (int i = 0; i < 256; ++i) { const float e = __expf(row[kb0 + i] - mx); row[kb0 + i] = e; sum += e; }
  sum += __shfl_xor(sum, 16, 32);
  const float inv = 1.f / sum;
  for (int i = 0; i < 256; ++i) row[kb0 + i] *= inv;
  __syncthreads();

  // ctx = P @ V   (V pre-transposed: Vt[hd_global][key])
#pragma unroll
  for (int nt = 0; nt < 4; ++nt) {
    v8f acc = {};
    for (int kc = 0; kc < 16; ++kc) {
      const int kb = kc * 32;
      const float* pr = sc + l15 * SCP + kb;
      v16bf pa;
#pragma unroll
      for (int i = 0; i < 8; ++i) {
        pa[i] = (__bf16)pr[half * 16 + i];
        pa[i + 8] = (__bf16)pr[half * 16 + 8 + i];
      }
      const __bf16* vr = Vt + (size_t)(h * 64 + nt * 16 + l15) * 512 + kb + half * 16;
      acc = WMMA_BF16(pa, ld_b16(vr), acc);
    }
#pragma unroll
    for (int r = 0; r < 8; ++r)
      ctx[(size_t)(t0 + r + half * 8) * 256 + h * 64 + nt * 16 + l15] = (__bf16)acc[r];
  }
}

// ---- exact top-512 of 4096 via rank counting (tie-break = lower index first) ----
__global__ __launch_bounds__(1024) void topk_kernel(const float* __restrict__ scores,
                                                    int* __restrict__ idx) {
  __shared__ float s[4096];
  for (int i = threadIdx.x; i < 4096; i += 1024) s[i] = scores[i];
  __syncthreads();
  for (int i = threadIdx.x; i < 4096; i += 1024) {
    const float v = s[i];
    int rank = 0;
    for (int j = 0; j < 4096; ++j) {
      const float u = s[j];
      rank += (u > v) || (u == v && j < i);
    }
    if (rank < 512) idx[rank] = i;
  }
}

static __device__ __forceinline__ float block_reduce_sum(float v, float* sbuf) {
  const int t = threadIdx.x;
  __syncthreads();
  sbuf[t] = v;
  __syncthreads();
  for (int s = 128; s > 0; s >>= 1) { if (t < s) sbuf[t] += sbuf[t + s]; __syncthreads(); }
  return sbuf[0];
}

__global__ __launch_bounds__(256) void gather_ln_kernel(
    const int* __restrict__ idx, const float* __restrict__ mem_keys,
    const float* __restrict__ g, const float* __restrict__ b, __bf16* __restrict__ out) {
  __shared__ float sbuf[256];
  const int slot = blockIdx.x;
  const int r = idx[slot];
  const float x = mem_keys[(size_t)r * 256 + threadIdx.x];
  const float s1 = block_reduce_sum(x, sbuf);
  const float s2 = block_reduce_sum(x * x, sbuf);
  const float mean = s1 / 256.f, var = s2 / 256.f - mean * mean;
  const float y = (x - mean) * rsqrtf(var + 1e-5f) * g[threadIdx.x] + b[threadIdx.x];
  out[(size_t)slot * 256 + threadIdx.x] = (__bf16)y;
}

// LN(2048) + exact GELU -> bf16
__global__ __launch_bounds__(256) void ln_gelu_kernel(
    const float* __restrict__ hpre, const float* __restrict__ g,
    const float* __restrict__ b, __bf16* __restrict__ hbf) {
  __shared__ float sbuf[256];
  const size_t base = (size_t)blockIdx.x * 2048;
  float vals[8], s1 = 0.f, s2 = 0.f;
#pragma unroll
  for (int i = 0; i < 8; ++i) {
    const float v = hpre[base + threadIdx.x + i * 256];
    vals[i] = v; s1 += v; s2 += v * v;
  }
  const float t1 = block_reduce_sum(s1, sbuf);
  const float t2 = block_reduce_sum(s2, sbuf);
  const float mean = t1 / 2048.f, var = t2 / 2048.f - mean * mean;
  const float inv = rsqrtf(var + 1e-5f);
#pragma unroll
  for (int i = 0; i < 8; ++i) {
    const int c = threadIdx.x + i * 256;
    float y = (vals[i] - mean) * inv * g[c] + b[c];
    y = 0.5f * y * (1.f + erff(y * 0.70710678118f));
    hbf[base + c] = (__bf16)y;
  }
}

// out = LN(X + gate*integrated)
__global__ __launch_bounds__(256) void final_kernel(
    const float* __restrict__ X, const float* __restrict__ gate,
    const float* __restrict__ intg, const float* __restrict__ g,
    const float* __restrict__ b, float* __restrict__ out) {
  __shared__ float sbuf[256];
  const size_t base = (size_t)blockIdx.x * 1024;
  float vals[4], s1 = 0.f, s2 = 0.f;
#pragma unroll
  for (int i = 0; i < 4; ++i) {
    const int c = threadIdx.x + i * 256;
    const float v = X[base + c] + gate[base + c] * intg[base + c];
    vals[i] = v; s1 += v; s2 += v * v;
  }
  const float t1 = block_reduce_sum(s1, sbuf);
  const float t2 = block_reduce_sum(s2, sbuf);
  const float mean = t1 / 1024.f, var = t2 / 1024.f - mean * mean;
  const float inv = rsqrtf(var + 1e-5f);
#pragma unroll
  for (int i = 0; i < 4; ++i) {
    const int c = threadIdx.x + i * 256;
    out[base + c] = (vals[i] - mean) * inv * g[c] + b[c];
  }
}

// f32 -> bf16, row-wise (dst may have larger row stride ldd)
__global__ void convert2d_kernel(const float* __restrict__ src, __bf16* __restrict__ dst,
                                 int cols, int ldd) {
  const int row = blockIdx.y;
  const int c = blockIdx.x * blockDim.x + threadIdx.x;
  if (c < cols) dst[(size_t)row * ldd + c] = (__bf16)src[(size_t)row * cols + c];
}

// folded query weight: qw[n][k] = sum_m wq[n][m]*qp_w[m][k]  (wq = in_proj_w rows 0..255)
__global__ __launch_bounds__(256) void qw_kernel(const float* __restrict__ in_proj_w,
                                                 const float* __restrict__ qp_w,
                                                 __bf16* __restrict__ qw) {
  const int n = blockIdx.y;
  const int k = blockIdx.x * 256 + threadIdx.x;
  float s = 0.f;
  for (int m = 0; m < 256; ++m) s += in_proj_w[n * 256 + m] * qp_w[(size_t)m * 1024 + k];
  qw[(size_t)n * 1024 + k] = (__bf16)s;
}
__global__ void qb_kernel(const float* __restrict__ in_proj_w, const float* __restrict__ qp_b,
                          const float* __restrict__ in_proj_b, float* __restrict__ qb) {
  const int n = threadIdx.x;
  float s = in_proj_b[n];
  for (int m = 0; m < 256; ++m) s += in_proj_w[n * 256 + m] * qp_b[m];
  qb[n] = s;
}

extern "C" void kernel_launch(void* const* d_in, const int* in_sizes, int n_in,
                              void* d_out, int out_size, void* d_ws, size_t ws_size,
                              hipStream_t stream) {
  (void)in_sizes; (void)n_in; (void)out_size;
  const float* X         = (const float*)d_in[0];   // [16384,1024]
  const float* mem_keys  = (const float*)d_in[1];   // [4096,256]
  const float* sel       = (const float*)d_in[2];   // [4096]
  const float* qp_w      = (const float*)d_in[3];   // [256,1024]
  const float* qp_b      = (const float*)d_in[4];   // [256]
  const float* in_proj_w = (const float*)d_in[5];   // [768,256]
  const float* in_proj_b = (const float*)d_in[6];   // [768]
  const float* out_w     = (const float*)d_in[7];   // [256,256]
  const float* out_b     = (const float*)d_in[8];
  const float* gate_w    = (const float*)d_in[9];   // [1024,1280]
  const float* gate_b    = (const float*)d_in[10];
  const float* int_w1    = (const float*)d_in[11];  // [2048,1280]
  const float* int_b1    = (const float*)d_in[12];
  const float* int_ln_g  = (const float*)d_in[13];
  const float* int_ln_b  = (const float*)d_in[14];
  const float* int_w2    = (const float*)d_in[15];  // [1024,2048]
  const float* int_b2    = (const float*)d_in[16];
  const float* ln1_g     = (const float*)d_in[17];
  const float* ln1_b     = (const float*)d_in[18];
  const float* ln2_g     = (const float*)d_in[19];
  const float* ln2_b     = (const float*)d_in[20];

  char* w = (char*)d_ws;
  size_t off = 0;
  auto alloc = [&](size_t bytes) -> char* {
    char* p = w + off;
    off = (off + bytes + 255) & ~(size_t)255;
    return p;
  };
  __bf16* cat      = (__bf16*)alloc((size_t)16384 * 1280 * 2);  // [X_bf | attn_out_bf]
  float*  hpre     = (float*)alloc((size_t)16384 * 2048 * 4);
  __bf16* hbf      = (__bf16*)alloc((size_t)16384 * 2048 * 2);
  float*  gateB    = (float*)alloc((size_t)16384 * 1024 * 4);
  __bf16* Qb       = (__bf16*)alloc((size_t)16384 * 256 * 2);
  __bf16* ctx      = (__bf16*)alloc((size_t)16384 * 256 * 2);
  __bf16* ln_mem   = (__bf16*)alloc((size_t)512 * 256 * 2);
  __bf16* Kmat     = (__bf16*)alloc((size_t)512 * 256 * 2);
  __bf16* vT       = (__bf16*)alloc((size_t)256 * 512 * 2);
  __bf16* wkv_bf   = (__bf16*)alloc((size_t)512 * 256 * 2);   // wk | wv
  __bf16* outw_bf  = (__bf16*)alloc((size_t)256 * 256 * 2);
  __bf16* gatew_bf = (__bf16*)alloc((size_t)1024 * 1280 * 2);
  __bf16* intw1_bf = (__bf16*)alloc((size_t)2048 * 1280 * 2);
  __bf16* intw2_bf = (__bf16*)alloc((size_t)1024 * 2048 * 2);
  __bf16* qw_bf    = (__bf16*)alloc((size_t)256 * 1024 * 2);
  float*  qb       = (float*)alloc(256 * 4);
  int*    idx      = (int*)alloc(512 * 4);
  float*  intg     = hpre;  // reuse hpre region after h_bf is produced
  if (ws_size < off) return;

  // weight precompute / conversions
  convert2d_kernel<<<dim3(1, 512), 256, 0, stream>>>(in_proj_w + 256 * 256, wkv_bf, 256, 256);
  convert2d_kernel<<<dim3(1, 256), 256, 0, stream>>>(out_w, outw_bf, 256, 256);
  convert2d_kernel<<<dim3(5, 1024), 256, 0, stream>>>(gate_w, gatew_bf, 1280, 1280);
  convert2d_kernel<<<dim3(5, 2048), 256, 0, stream>>>(int_w1, intw1_bf, 1280, 1280);
  convert2d_kernel<<<dim3(8, 1024), 256, 0, stream>>>(int_w2, intw2_bf, 2048, 2048);
  convert2d_kernel<<<dim3(4, 16384), 256, 0, stream>>>(X, cat, 1024, 1280);
  qw_kernel<<<dim3(4, 256), 256, 0, stream>>>(in_proj_w, qp_w, qw_bf);
  qb_kernel<<<1, 256, 0, stream>>>(in_proj_w, qp_b, in_proj_b, qb);

  // memory selection + LN
  topk_kernel<<<1, 1024, 0, stream>>>(sel, idx);
  gather_ln_kernel<<<512, 256, 0, stream>>>(idx, mem_keys, ln1_g, ln1_b, ln_mem);

  // K, V^T projections  (M=512, N=256, K=256)
  gemm_bf16_kernel<1><<<dim3(2, 8), 256, 0, stream>>>(ln_mem, 256, wkv_bf, 256,
                                                      in_proj_b + 256, Kmat, 256, 256);
  gemm_bf16_kernel<4><<<dim3(2, 8), 256, 0, stream>>>(ln_mem, 256, wkv_bf + 256 * 256, 256,
                                                      in_proj_b + 512, vT, 512, 256);
  // Q = 0.125 * (X @ qw^T + qb)   (M=16384, N=256, K=1024)
  gemm_bf16_kernel<2><<<dim3(2, 256), 256, 0, stream>>>(cat, 1280, qw_bf, 1024, qb, Qb, 256, 1024);

  // fused attention
  attn_kernel<<<dim3(1024, 4), 32, 0, stream>>>(Qb, Kmat, vT, ctx);

  // attn_out -> cat[:, 1024:1280]
  gemm_bf16_kernel<1><<<dim3(2, 256), 256, 0, stream>>>(ctx, 256, outw_bf, 256, out_b,
                                                        cat + 1024, 1280, 256);
  // gate = sigmoid(cat @ gate_w^T + gate_b)
  gemm_bf16_kernel<3><<<dim3(8, 256), 256, 0, stream>>>(cat, 1280, gatew_bf, 1280, gate_b,
                                                        gateB, 1024, 1280);
  // h_pre = cat @ int_w1^T + int_b1
  gemm_bf16_kernel<0><<<dim3(16, 256), 256, 0, stream>>>(cat, 1280, intw1_bf, 1280, int_b1,
                                                         hpre, 2048, 1280);
  ln_gelu_kernel<<<16384, 256, 0, stream>>>(hpre, int_ln_g, int_ln_b, hbf);
  // integrated = h @ int_w2^T + int_b2  (aliases hpre region; hpre no longer needed)
  gemm_bf16_kernel<0><<<dim3(8, 256), 256, 0, stream>>>(hbf, 2048, intw2_bf, 2048, int_b2,
                                                        intg, 1024, 2048);
  final_kernel<<<16384, 256, 0, stream>>>(X, gateB, intg, ln2_g, ln2_b, (float*)d_out);
}
